// TransformerBlock_4733053960626
// MI455X (gfx1250) — matrix operand order repaired and checked by reading, not compiled, not run on hardware
//
#include <hip/hip_runtime.h>
#include <math.h>

// ---------------- constants ----------------
#define TOK   2048          // B*N tokens
#define NSEQ  1024          // sequence length per batch
#define DMODEL 512

typedef __attribute__((ext_vector_type(16))) __bf16        v16bf;
typedef __attribute__((ext_vector_type(8)))  unsigned int  v8u;
typedef __attribute__((ext_vector_type(8)))  float         v8f;

// pack two fp32 -> bf16 pair (a -> low 16, b -> high 16); round-to-nearest
__device__ __forceinline__ unsigned int pack_bf16(float a, float b) {
  unsigned int ua = __float_as_uint(a) + 0x8000u;
  unsigned int ub = __float_as_uint(b) + 0x8000u;
  return __builtin_amdgcn_perm(ub, ua, 0x07060302u);  // {ub[31:16], ua[31:16]}
}
__device__ __forceinline__ float gelu_f(float x) {
  const float c = 0.7978845608028654f;       // sqrt(2/pi)  (jax tanh-approx gelu)
  return 0.5f * x * (1.0f + tanhf(c * (x + 0.044715f * x * x * x)));
}

// ---------------- WMMA GEMM ----------------
// One wave (32 threads) computes a 32x64 tile of C = act(A[MxK] @ B + bias).
// A row-major (lda); B row-major KxN (ldb) or, if TRANSB, NxK (ldb).
// Requires M%32==0, N%64==0, K%32==0.
template <int TRANSB>
__global__ void __launch_bounds__(32, 2)
k_wmma_gemm_t(const float* __restrict__ A, int lda,
              const float* __restrict__ Bm, int ldb,
              float* __restrict__ C, int ldc, int K,
              const float* __restrict__ bias,
              int act, int accum)
{
  const int lane = threadIdx.x & 31;
  const int half = lane >> 4, l16 = lane & 15;
  const int n0 = blockIdx.x * 64, m0 = blockIdx.y * 32;

  v8f acc[2][4] = {};

  // A fragment rows for this lane (16-bit A 16x32: M = l16, elem i -> K = half*16 + i)
  const float* Arow[2];
  Arow[0] = A + (size_t)(m0 + l16) * lda + half * 16;
  Arow[1] = Arow[0] + (size_t)16 * lda;

  for (int k0 = 0; k0 < K; k0 += 32) {
    v8u af[2], bfr[4];
#pragma unroll
    for (int t = 0; t < 2; ++t) {
      const float* p = Arow[t] + k0;
      const float4 f0 = *(const float4*)(p + 0);
      const float4 f1 = *(const float4*)(p + 4);
      const float4 f2 = *(const float4*)(p + 8);
      const float4 f3 = *(const float4*)(p + 12);
      af[t][0] = pack_bf16(f0.x, f0.y); af[t][1] = pack_bf16(f0.z, f0.w);
      af[t][2] = pack_bf16(f1.x, f1.y); af[t][3] = pack_bf16(f1.z, f1.w);
      af[t][4] = pack_bf16(f2.x, f2.y); af[t][5] = pack_bf16(f2.z, f2.w);
      af[t][6] = pack_bf16(f3.x, f3.y); af[t][7] = pack_bf16(f3.z, f3.w);
    }
    // B fragments: 32x16 bf16 tiles, N = l16, elem i -> K = half*16 + i
    if (TRANSB) {
#pragma unroll
      for (int t = 0; t < 4; ++t) {
        const float* p = Bm + (size_t)(n0 + t * 16 + l16) * ldb + k0 + half * 16;
        const float4 f0 = *(const float4*)(p + 0);
        const float4 f1 = *(const float4*)(p + 4);
        const float4 f2 = *(const float4*)(p + 8);
        const float4 f3 = *(const float4*)(p + 12);
        bfr[t][0] = pack_bf16(f0.x, f0.y); bfr[t][1] = pack_bf16(f0.z, f0.w);
        bfr[t][2] = pack_bf16(f1.x, f1.y); bfr[t][3] = pack_bf16(f1.z, f1.w);
        bfr[t][4] = pack_bf16(f2.x, f2.y); bfr[t][5] = pack_bf16(f2.z, f2.w);
        bfr[t][6] = pack_bf16(f3.x, f3.y); bfr[t][7] = pack_bf16(f3.z, f3.w);
      }
    } else {
#pragma unroll
      for (int t = 0; t < 4; ++t) {
        const float* p = Bm + (size_t)(k0 + half * 16) * ldb + (n0 + t * 16 + l16);
#pragma unroll
        for (int i = 0; i < 8; ++i)
          bfr[t][i] = pack_bf16(p[(size_t)(2 * i) * ldb], p[(size_t)(2 * i + 1) * ldb]);
      }
    }
#pragma unroll
    for (int i = 0; i < 2; ++i)
#pragma unroll
      for (int j = 0; j < 4; ++j)
        acc[i][j] = __builtin_amdgcn_wmma_f32_16x16x32_bf16(
            false, __builtin_bit_cast(v16bf, af[i]),
            false, __builtin_bit_cast(v16bf, bfr[j]),
            (short)0, acc[i][j], false, false);
  }

  // ---- epilogue: C VGPR r -> M = half*8 + r, N = l16
#pragma unroll
  for (int i = 0; i < 2; ++i)
#pragma unroll
    for (int j = 0; j < 4; ++j) {
      const int n = n0 + j * 16 + l16;
      const float bv = bias ? bias[n] : 0.0f;
#pragma unroll
      for (int r = 0; r < 8; ++r) {
        int m = m0 + i * 16 + half * 8 + r;
        float v = acc[i][j][r] + bv;
        if (act) v = gelu_f(v);
        float* cp = C + (size_t)m * ldc + n;
        if (accum) v += *cp;
        *cp = v;
      }
    }
}

// ---------------- elementwise / reduction kernels ----------------
__global__ void k_rms(const float* __restrict__ x, const float* __restrict__ g,
                      const float* __restrict__ res, float* __restrict__ y, int C)
{
  __shared__ float red[256];
  const int r = blockIdx.x, tid = threadIdx.x;
  const float* xr = x + (size_t)r * C;
  float s = 0.f;
  for (int j = tid; j < C; j += 256) { float v = xr[j]; s += v * v; }
  red[tid] = s; __syncthreads();
  for (int o = 128; o > 0; o >>= 1) { if (tid < o) red[tid] += red[tid + o]; __syncthreads(); }
  float norm = sqrtf(red[0] / (float)C);
  float inv = 1.0f / fmaxf(norm, 1e-8f);
  float* yr = y + (size_t)r * C;
  const float* rr = res ? res + (size_t)r * C : nullptr;
  for (int j = tid; j < C; j += 256) {
    float v = xr[j] * inv * g[j];
    if (rr) v += rr[j];
    yr[j] = v;
  }
}

__global__ void k_softmax(float* __restrict__ x, int C, float scale)
{
  __shared__ float red[256];
  const int r = blockIdx.x, tid = threadIdx.x;
  float* row = x + (size_t)r * C;
  float m = -3.0e38f;
  for (int j = tid; j < C; j += 256) m = fmaxf(m, row[j] * scale);
  red[tid] = m; __syncthreads();
  for (int o = 128; o > 0; o >>= 1) { if (tid < o) red[tid] = fmaxf(red[tid], red[tid + o]); __syncthreads(); }
  m = red[0]; __syncthreads();
  float s = 0.f;
  for (int j = tid; j < C; j += 256) { float e = expf(row[j] * scale - m); row[j] = e; s += e; }
  red[tid] = s; __syncthreads();
  for (int o = 128; o > 0; o >>= 1) { if (tid < o) red[tid] += red[tid + o]; __syncthreads(); }
  float inv = 1.0f / red[0];
  for (int j = tid; j < C; j += 256) row[j] *= inv;
}

__global__ void k_gelu(const float* x, float* y, int n)
{ int i = blockIdx.x * 256 + threadIdx.x; if (i < n) y[i] = gelu_f(x[i]); }

// y = u[:, :C] * gelu(u[:, C:]) (+ add)
__global__ void k_geglu_add(const float* __restrict__ u, const float* __restrict__ add,
                            float* __restrict__ y, int n, int C)
{
  int i = blockIdx.x * 256 + threadIdx.x; if (i >= n) return;
  int t = i / C, c = i % C;
  float a = u[(size_t)t * 2 * C + c];
  float g = u[(size_t)t * 2 * C + C + c];
  float v = a * gelu_f(g);
  if (add) v += add[i];
  y[i] = v;
}

__global__ void k_add2(const float* a, const float* b, float* y, int n)
{ int i = blockIdx.x * 256 + threadIdx.x; if (i < n) y[i] = a[i] + b[i]; }
__global__ void k_add3(const float* a, const float* b, const float* c, float* y, int n)
{ int i = blockIdx.x * 256 + threadIdx.x; if (i < n) y[i] = a[i] + b[i] + c[i]; }

__global__ void k_gru(const float* __restrict__ gi, const float* __restrict__ gh,
                      const float* __restrict__ hf, float* __restrict__ out, int n)
{
  int i = blockIdx.x * 256 + threadIdx.x; if (i >= n) return;
  int t = i / DMODEL, c = i % DMODEL;
  size_t b = (size_t)t * 3 * DMODEL;
  float ir = gi[b + c], iz = gi[b + DMODEL + c], ig = gi[b + 2 * DMODEL + c];
  float hr = gh[b + c], hz = gh[b + DMODEL + c], hg = gh[b + 2 * DMODEL + c];
  float rg = 1.0f / (1.0f + expf(-(ir + hr)));
  float z  = 1.0f / (1.0f + expf(-(iz + hz)));
  float g  = tanhf(ig + rg * hg);
  float h  = hf[i];
  out[i] = (1.0f - z) * g + z * h;
}

// y = (hop2[:, :512]*g + x) * gelu(hop2[:, 512:])
__global__ void k_mncomb(const float* __restrict__ hop2, const float* __restrict__ x,
                         const float* __restrict__ g, float* __restrict__ y, int n)
{
  int i = blockIdx.x * 256 + threadIdx.x; if (i >= n) return;
  int t = i / DMODEL, c = i % DMODEL;
  float go = hop2[(size_t)t * 1024 + c];
  float ga = hop2[(size_t)t * 1024 + DMODEL + c];
  y[i] = (go * g[0] + x[i]) * gelu_f(ga);
}

__global__ void k_posenc(float* x, int n)
{
  int i = blockIdx.x * 256 + threadIdx.x; if (i >= n) return;
  int t = i / DMODEL, d = i % DMODEL;
  int pos = t % NSEQ, i2 = d >> 1;
  float ang = (float)pos * expf((float)i2 * -0.035977892078032f); // -ln(10000)*2/512
  x[i] += (d & 1) ? cosf(ang) : sinf(ang);
}

// y[t,oc] = gelu(bias[oc] + sum_{kw,ic} x[b, n+kw-1, ic] * w[kw,ic,oc]); in=512, out=256, k=3, SAME
__global__ void k_conv3(const float* __restrict__ x, const float* __restrict__ w,
                        const float* __restrict__ b, float* __restrict__ y, int n)
{
  int i = blockIdx.x * 256 + threadIdx.x; if (i >= n) return;
  int t = i / 256, oc = i % 256;
  int bi = t / NSEQ, nn0 = t % NSEQ;
  float acc = b[oc];
  for (int kw = 0; kw < 3; ++kw) {
    int nn = nn0 + kw - 1;
    if ((unsigned)nn < (unsigned)NSEQ) {
      const float* row = x + ((size_t)(bi * NSEQ + nn)) * DMODEL;
      const float* wp  = w + (size_t)kw * DMODEL * 256 + oc;
      for (int ic = 0; ic < DMODEL; ++ic) acc += row[ic] * wp[(size_t)ic * 256];
    }
  }
  y[i] = gelu_f(acc);
}

// left[t, c] += right[t, c] for c<256 (left has 2048 cols)
__global__ void k_padadd2048(float* __restrict__ left, const float* __restrict__ right, int n)
{
  int i = blockIdx.x * 256 + threadIdx.x; if (i >= n) return;
  int t = i / 256, c = i % 256;
  left[(size_t)t * 2048 + c] += right[i];
}

// depthwise conv, k=9, SAME, channels=2048
__global__ void k_dwconv9(const float* __restrict__ x, const float* __restrict__ w,
                          float* __restrict__ y, int n)
{
  int i = blockIdx.x * 256 + threadIdx.x; if (i >= n) return;
  int t = i / 2048, c = i % 2048;
  int bi = t / NSEQ, nn0 = t % NSEQ;
  float acc = 0.f;
  for (int j = 0; j < 9; ++j) {
    int nn = nn0 + j - 4;
    if ((unsigned)nn < (unsigned)NSEQ)
      acc += x[((size_t)(bi * NSEQ + nn)) * 2048 + c] * w[j * 2048 + c];
  }
  y[i] = acc;
}

// out[t,c] = base[t,c] + (c<256 ? small[t,c] : 0)   (512 cols)
__global__ void k_addpad(const float* __restrict__ base, const float* __restrict__ small,
                         float* __restrict__ out, int n)
{
  int i = blockIdx.x * 256 + threadIdx.x; if (i >= n) return;
  int t = i / DMODEL, c = i % DMODEL;
  float v = base[i];
  if (c < 256) v += small[(size_t)t * 256 + c];
  out[i] = v;
}

// ---------------- PKM top-k / gather ----------------
// dots: (T, 4 heads, 2 halves, 128). out: (T, 256). One block (128 thr) per token.
__global__ void k_pkm_topk(const float* __restrict__ dots, const float* __restrict__ values,
                           float* __restrict__ out)
{
  __shared__ float sv[128];
  __shared__ float s1[32], s2[32];
  __shared__ int   i1[32], i2[32];
  __shared__ float comb[1024];
  __shared__ float sel_s[32];
  __shared__ int   sel_i[32];
  __shared__ float acc[256];
  __shared__ float rv[128];
  __shared__ int   ri[128];

  const int t = blockIdx.x, tid = threadIdx.x;
  acc[tid] = 0.f; acc[tid + 128] = 0.f;
  __syncthreads();

  for (int h = 0; h < 4; ++h) {
    for (int p = 0; p < 2; ++p) {
      float* S = p ? s2 : s1; int* I = p ? i2 : i1;
      sv[tid] = dots[(size_t)t * 1024 + (h * 2 + p) * 128 + tid];
      __syncthreads();
      for (int k = 0; k < 32; ++k) {
        rv[tid] = sv[tid]; ri[tid] = tid; __syncthreads();
        for (int o = 64; o > 0; o >>= 1) {
          if (tid < o) {
            float a = rv[tid], b = rv[tid + o];
            int ia = ri[tid], ib = ri[tid + o];
            if (b > a || (b == a && ib < ia)) { rv[tid] = b; ri[tid] = ib; }
          }
          __syncthreads();
        }
        if (tid == 0) { S[k] = rv[0]; I[k] = ri[0]; sv[ri[0]] = -3.0e38f; }
        __syncthreads();
      }
    }
    for (int j = tid; j < 1024; j += 128) comb[j] = s1[j >> 5] + s2[j & 31];
    __syncthreads();
    for (int k = 0; k < 32; ++k) {
      float bv = -3.0e38f; int bi = 0x7fffffff;
      for (int j = tid; j < 1024; j += 128) {
        float v = comb[j];
        if (v > bv || (v == bv && j < bi)) { bv = v; bi = j; }
      }
      rv[tid] = bv; ri[tid] = bi; __syncthreads();
      for (int o = 64; o > 0; o >>= 1) {
        if (tid < o) {
          float a = rv[tid], b = rv[tid + o];
          int ia = ri[tid], ib = ri[tid + o];
          if (b > a || (b == a && ib < ia)) { rv[tid] = b; ri[tid] = ib; }
        }
        __syncthreads();
      }
      if (tid == 0) { sel_s[k] = rv[0]; sel_i[k] = ri[0]; comb[ri[0]] = -3.0e38f; }
      __syncthreads();
    }
    if (tid == 0) {
      float m = sel_s[0];
      for (int k = 1; k < 32; ++k) m = fmaxf(m, sel_s[k]);
      float s = 0.f;
      for (int k = 0; k < 32; ++k) { float e = expf(sel_s[k] - m); sel_s[k] = e; s += e; }
      float inv = 1.0f / s;
      for (int k = 0; k < 32; ++k) sel_s[k] *= inv;
    }
    __syncthreads();
    for (int k = 0; k < 32; ++k) {
      int loc = sel_i[k];
      long long row = (long long)(i1[loc >> 5] * 128 + i2[loc & 31]) * 256;
      float w = sel_s[k];
      acc[tid]       += w * values[row + tid];
      acc[tid + 128] += w * values[row + tid + 128];
    }
    __syncthreads();
  }
  out[(size_t)t * 256 + tid]       = acc[tid];
  out[(size_t)t * 256 + tid + 128] = acc[tid + 128];
}

// ---------------- host orchestration ----------------
static inline const float* FP(void* const* din, int i) { return (const float*)din[i]; }
static inline int cdiv(int a, int b) { return (a + b - 1) / b; }

static void gemm(hipStream_t st, const float* A, int lda, const float* B, int ldb,
                 float* C, int ldc, int M, int N, int K,
                 const float* bias, int act, int transB, int accum)
{
  dim3 g(N / 64, M / 32, 1), blk(32, 1, 1);
  if (transB)
    k_wmma_gemm_t<1><<<g, blk, 0, st>>>(A, lda, B, ldb, C, ldc, K, bias, act, accum);
  else
    k_wmma_gemm_t<0><<<g, blk, 0, st>>>(A, lda, B, ldb, C, ldc, K, bias, act, accum);
}

struct Bufs {
  float *bigA, *bigB;                 // (T,2048) each
  float *a, *b, *c, *d, *e;           // (T,512) each
  float *h1024;                       // (T,1024)
  float *p256a, *p256b;               // (T,256)
  float *h64;                         // (T,64)
};

static void run_hop(hipStream_t st, const float* x, int dim, int dout,
                    const float* wq, const float* Km, const float* Vm, const float* wout,
                    float* out, float* qb, float* sc, float* av)
{
  gemm(st, x, dim, wq, dim, qb, dim, TOK, dim, dim, nullptr, 0, 0, 0);
  gemm(st, qb, dim, Km, dim, sc, 64, TOK, 64, dim, nullptr, 0, 1, 0);
  k_softmax<<<TOK, 256, 0, st>>>(sc, 64, 1.0f / sqrtf((float)dim));
  gemm(st, sc, 64, Vm, dim, av, dim, TOK, dim, 64, nullptr, 0, 0, 0);
  gemm(st, av, dim, wout, dout, out, dout, TOK, dout, dim, nullptr, 0, 0, 0);
}

static void run_pkm(hipStream_t st, void* const* din, int base,
                    const float* x, float* out, Bufs& B)
{
  const float* in_w = FP(din, base), *in_b = FP(din, base + 1);
  const float* wq = FP(din, base + 2), *keys = FP(din, base + 3);
  const float* values = FP(din, base + 4);
  const float* out_w = FP(din, base + 5), *out_b = FP(din, base + 6);
  float* h = B.bigA;                        // (T,256)
  float* q = B.bigA + (size_t)TOK * 256;    // (T,1024)
  float* dots = B.bigB;                     // (T,1024)
  float* sum = B.bigB + (size_t)TOK * 1024; // (T,256)
  gemm(st, x, 512, in_w, 256, h, 256, TOK, 256, 512, in_b, 1, 0, 0);
  gemm(st, h, 256, wq, 1024, q, 1024, TOK, 1024, 256, nullptr, 0, 0, 0);
  for (int hh = 0; hh < 4; ++hh)
    for (int p = 0; p < 2; ++p) {
      int bi = hh * 2 + p;
      gemm(st, q + bi * 128, 1024, keys + hh * 32768 + p * 128, 256,
           dots + bi * 128, 1024, TOK, 128, 128, nullptr, 0, 1, 0);
    }
  k_pkm_topk<<<TOK, 128, 0, st>>>(dots, values, sum);
  k_gelu<<<cdiv(TOK * 256, 256), 256, 0, st>>>(sum, sum, TOK * 256);
  gemm(st, sum, 256, out_w, 512, out, 512, TOK, 512, 256, out_b, 1, 0, 0);
}

struct EvIdx {
  int n0, n1, n2, n3, geglu_w, geglu_b, left_w, left_b, right_w, right_b,
      mid_g, sep_dw, sep_pw, sep_b;
  int a0, a1, a2, a3, a4, a5, a6, a7;   // attn params (hop or self)
  int f0, f1, f2, f3, f4, f5;           // ffd (+geglu)
  int pkm;                              // pkm base
  int attn_self, ffd_geglu;
};

static void run_evolved(hipStream_t st, void* const* din, const EvIdx& e, Bufs& B)
{
  const int NW = TOK * 512;
  // glu_out = geglu(rms(x)) + x     (x in B.a)
  k_rms<<<TOK, 256, 0, st>>>(B.a, FP(din, e.n0), nullptr, B.b, 512);
  gemm(st, B.b, 512, FP(din, e.geglu_w), 1024, B.h1024, 1024, TOK, 1024, 512, FP(din, e.geglu_b), 0, 0, 0);
  k_geglu_add<<<cdiv(NW, 256), 256, 0, st>>>(B.h1024, B.a, B.c, NW, 512);      // glu_out -> c
  k_rms<<<TOK, 256, 0, st>>>(B.c, FP(din, e.n1), nullptr, B.b, 512);           // nx -> b
  // left (T,2048) + padded right (T,256)
  gemm(st, B.b, 512, FP(din, e.left_w), 2048, B.bigA, 2048, TOK, 2048, 512, FP(din, e.left_b), 1, 0, 0);
  k_conv3<<<cdiv(TOK * 256, 256), 256, 0, st>>>(B.b, FP(din, e.right_w), FP(din, e.right_b), B.p256a, TOK * 256);
  k_padadd2048<<<cdiv(TOK * 256, 256), 256, 0, st>>>(B.bigA, B.p256a, TOK * 256);
  k_rms<<<TOK, 256, 0, st>>>(B.bigA, FP(din, e.mid_g), nullptr, B.bigB, 2048); // mid -> bigB
  k_dwconv9<<<cdiv(TOK * 2048, 256), 256, 0, st>>>(B.bigB, FP(din, e.sep_dw), B.bigA, TOK * 2048);
  gemm(st, B.bigA, 2048, FP(din, e.sep_pw), 256, B.p256a, 256, TOK, 256, 2048, FP(din, e.sep_b), 0, 0, 0);
  k_addpad<<<cdiv(NW, 256), 256, 0, st>>>(B.c, B.p256a, B.d, NW);              // mid_out -> d
  k_rms<<<TOK, 256, 0, st>>>(B.d, FP(din, e.n2), nullptr, B.b, 512);           // nx2 -> b
  // ---- attention -> e
  if (!e.attn_self) {
    gemm(st, B.b, 512, FP(din, e.a0), 256, B.p256a, 256, TOK, 256, 512, FP(din, e.a1), 0, 0, 0);
    run_hop(st, B.p256a, 256, 256, FP(din, e.a2), FP(din, e.a3), FP(din, e.a4), FP(din, e.a5),
            B.p256b, B.bigA, B.bigA + (size_t)TOK * 256, B.bigA + (size_t)TOK * 256 + (size_t)TOK * 64);
    gemm(st, B.p256b, 256, FP(din, e.a6), 512, B.e, 512, TOK, 512, 256, FP(din, e.a7), 0, 0, 0);
  } else {
    float* q = B.bigA;
    float* k = B.bigA + (size_t)TOK * 512;
    float* v = B.bigA + (size_t)TOK * 1024;
    float* o = B.bigA + (size_t)TOK * 1536;
    gemm(st, B.b, 512, FP(din, e.a0), 512, q, 512, TOK, 512, 512, nullptr, 0, 0, 0);
    gemm(st, B.b, 512, FP(din, e.a1), 512, k, 512, TOK, 512, 512, nullptr, 0, 0, 0);
    gemm(st, B.b, 512, FP(din, e.a2), 512, v, 512, TOK, 512, 512, nullptr, 0, 0, 0);
    for (int bb = 0; bb < 2; ++bb)
      for (int h = 0; h < 8; ++h) {
        const float* Aq = q + (size_t)bb * NSEQ * 512 + h * 64;
        float* sc = B.bigB;                    // (1024, 3072)
        gemm(st, Aq, 512, FP(din, e.a3) + (size_t)h * 2048 * 64, 64, sc, 3072, NSEQ, 2048, 64, nullptr, 0, 1, 0);
        gemm(st, Aq, 512, k + (size_t)bb * NSEQ * 512 + h * 64, 512, sc + 2048, 3072, NSEQ, 1024, 64, nullptr, 0, 1, 0);
        k_softmax<<<NSEQ, 256, 0, st>>>(sc, 3072, 0.125f);
        float* oh = o + (size_t)bb * NSEQ * 512 + h * 64;
        gemm(st, sc, 3072, FP(din, e.a4) + (size_t)h * 2048 * 64, 64, oh, 512, NSEQ, 64, 2048, nullptr, 0, 0, 0);
        gemm(st, sc + 2048, 3072, v + (size_t)bb * NSEQ * 512 + h * 64, 512, oh, 512, NSEQ, 64, 1024, nullptr, 0, 0, 1);
      }
    gemm(st, o, 512, FP(din, e.a5), 512, B.e, 512, TOK, 512, 512, FP(din, e.a6), 0, 0, 0);
  }
  // ---- PKM -> a (input x is dead), att = e + a + d -> e
  run_pkm(st, din, e.pkm, B.b, B.a, B);
  k_add3<<<cdiv(NW, 256), 256, 0, st>>>(B.e, B.a, B.d, B.e, NW);
  k_rms<<<TOK, 256, 0, st>>>(B.e, FP(din, e.n3), nullptr, B.b, 512);           // nx3 -> b
  // ---- feed-forward -> a, plus residual e
  if (!e.ffd_geglu) {
    gemm(st, B.b, 512, FP(din, e.f0), 2048, B.bigA, 2048, TOK, 2048, 512, FP(din, e.f1), 1, 0, 0);
    gemm(st, B.bigA, 2048, FP(din, e.f2), 512, B.a, 512, TOK, 512, 2048, FP(din, e.f3), 1, 0, 0);
    k_add2<<<cdiv(NW, 256), 256, 0, st>>>(B.a, B.e, B.a, NW);
  } else {
    gemm(st, B.b, 512, FP(din, e.f0), 2048, B.bigA, 2048, TOK, 2048, 512, FP(din, e.f1), 1, 0, 0);
    gemm(st, B.bigA, 2048, FP(din, e.f2), 512, B.c, 512, TOK, 512, 2048, FP(din, e.f3), 1, 0, 0);
    gemm(st, B.c, 512, FP(din, e.f4), 1024, B.h1024, 1024, TOK, 1024, 512, FP(din, e.f5), 0, 0, 0);
    k_geglu_add<<<cdiv(NW, 256), 256, 0, st>>>(B.h1024, B.e, B.a, NW, 512);
  }
}

static void run_memnorm(hipStream_t st, void* const* din, int base,
                        const float* x, const float* src, float* out, Bufs& B)
{
  const int NW = TOK * 512;
  gemm(st, x,   512, FP(din, base + 0), 1536, B.bigA, 1536, TOK, 1536, 512, FP(din, base + 2), 0, 0, 0);
  gemm(st, src, 512, FP(din, base + 1), 1536, B.bigB, 1536, TOK, 1536, 512, FP(din, base + 3), 0, 0, 0);
  k_gru<<<cdiv(NW, 256), 256, 0, st>>>(B.bigA, B.bigB, src, B.b, NW);
  run_hop(st, B.b, 512, 1024, FP(din, base + 4), FP(din, base + 5), FP(din, base + 6), FP(din, base + 7),
          B.h1024, B.bigA, B.bigA + (size_t)TOK * 512, B.bigA + (size_t)TOK * 512 + (size_t)TOK * 64);
  k_mncomb<<<cdiv(NW, 256), 256, 0, st>>>(B.h1024, x, FP(din, base + 8), B.c, NW);
  k_rms<<<TOK, 256, 0, st>>>(B.c, FP(din, base + 9), src, out, 512);
}

extern "C" void kernel_launch(void* const* d_in, const int* in_sizes, int n_in,
                              void* d_out, int out_size, void* d_ws, size_t ws_size,
                              hipStream_t stream)
{
  (void)in_sizes; (void)n_in; (void)out_size; (void)ws_size;
  const int NW = TOK * 512;
  const float* src = FP(d_in, 0);

  // workspace arena
  float* W = (float*)d_ws;
  size_t off = 0;
  auto alloc = [&](size_t n) { float* p = W + off; off += n; return p; };
  Bufs B;
  B.bigA  = alloc((size_t)TOK * 2048);
  B.bigB  = alloc((size_t)TOK * 2048);
  B.a     = alloc((size_t)TOK * 512);
  B.b     = alloc((size_t)TOK * 512);
  B.c     = alloc((size_t)TOK * 512);
  B.d     = alloc((size_t)TOK * 512);
  B.e     = alloc((size_t)TOK * 512);
  B.h1024 = alloc((size_t)TOK * 1024);
  B.p256a = alloc((size_t)TOK * 256);
  B.p256b = alloc((size_t)TOK * 256);
  B.h64   = alloc((size_t)TOK * 64);

  // ---- stage 1: out = ffd1(norm) + pkm1(norm) + geglu1(norm)
  k_rms<<<TOK, 256, 0, stream>>>(src, FP(d_in, 1), nullptr, B.a, 512);
  gemm(stream, B.a, 512, FP(d_in, 2), 2048, B.bigA, 2048, TOK, 2048, 512, FP(d_in, 3), 1, 0, 0);
  gemm(stream, B.bigA, 2048, FP(d_in, 4), 512, B.b, 512, TOK, 512, 2048, FP(d_in, 5), 1, 0, 0);
  run_pkm(stream, d_in, 8, B.a, B.c, B);
  gemm(stream, B.a, 512, FP(d_in, 6), 1024, B.h1024, 1024, TOK, 1024, 512, FP(d_in, 7), 0, 0, 0);
  k_geglu_add<<<cdiv(NW, 256), 256, 0, stream>>>(B.h1024, nullptr, B.d, NW, 512);
  k_add3<<<cdiv(NW, 256), 256, 0, stream>>>(B.b, B.c, B.d, B.a, NW);

  // ---- stage 2: evolved (hopfield attention + plain ffd)
  EvIdx e1 = {15, 16, 17, 18, 19, 20, 21, 22, 23, 24, 25, 26, 27, 28,
              29, 30, 31, 32, 33, 34, 35, 36,
              37, 38, 39, 40, -1, -1,
              41, /*attn_self=*/0, /*ffd_geglu=*/0};
  run_evolved(stream, d_in, e1, B);

  // ---- stage 3: memnorm gate1
  run_memnorm(stream, d_in, 48, B.a, src, B.a, B);

  // ---- stage 4: positional encoding
  k_posenc<<<cdiv(NW, 256), 256, 0, stream>>>(B.a, NW);

  // ---- stage 5: evolved (self-attention with memory KV + ffd_geglu)
  EvIdx e2 = {68, 69, 70, 71, 72, 73, 74, 75, 76, 77, 78, 79, 80, 81,
              82, 83, 84, 85, 86, 87, 88, -1,
              89, 90, 91, 92, 93, 94,
              95, /*attn_self=*/1, /*ffd_geglu=*/1};
  run_evolved(stream, d_in, e2, B);

  // ---- stage 6: memnorm gate2 -> output
  run_memnorm(stream, d_in, 58, B.a, src, (float*)d_out, B);
}
